// UnionRGCNLayer_63471026700599
// MI455X (gfx1250) — matrix operand order repaired and checked by reading, not compiled, not run on hardware
//
#include <hip/hip_runtime.h>
#include <hip/hip_bf16.h>

typedef __attribute__((ext_vector_type(16))) __bf16 v16bf;
typedef __attribute__((ext_vector_type(8)))  float  v8f;

#define DIM     128
#define NTILE_N 8     // 128 / 16 output column tiles
#define KSTEPS  4     // 128 / 32 WMMA K steps
#define SWZ_ELEMS (3 * NTILE_N * KSTEPS * 32 * 16)   // 49152 bf16 = 96 KB
#define SWZ_VCHUNKS (SWZ_ELEMS / 16)                 // 3072 x 32-byte chunks

// ---------------------------------------------------------------------------
// Kernel 1: per-edge scatter.  acc[dst] += h[src] + rel[etype]*time[etime]
// One wave32 per edge; lane l owns float4 l of the 128-float row (128/4==32).
// h (25.6MB) + acc (25.6MB) are L2-resident on MI455X (192MB L2), so the
// random gather/atomic traffic never leaves L2.
// ---------------------------------------------------------------------------
__global__ void edge_scatter_kernel(const float* __restrict__ h,
                                    const float* __restrict__ rel,
                                    const float* __restrict__ tim,
                                    const int*   __restrict__ src,
                                    const int*   __restrict__ dst,
                                    const int*   __restrict__ etype,
                                    const int*   __restrict__ etime,
                                    float*       __restrict__ acc,
                                    int*         __restrict__ indeg,
                                    int E)
{
    const int tid    = blockIdx.x * blockDim.x + threadIdx.x;
    const int lane   = tid & 31;
    const int wave   = tid >> 5;
    const int nwaves = (gridDim.x * blockDim.x) >> 5;

    for (int e = wave; e < E; e += nwaves) {
        const int s = src[e];
        const int d = dst[e];
        const int r = etype[e];
        const int t = etime[e];

        const float4 hv = ((const float4*)(h   + (size_t)s * DIM))[lane];
        const float4 rv = ((const float4*)(rel + (size_t)r * DIM))[lane];
        const float4 tv = ((const float4*)(tim + (size_t)t * DIM))[lane];

        float4 v;
        v.x = fmaf(rv.x, tv.x, hv.x);
        v.y = fmaf(rv.y, tv.y, hv.y);
        v.z = fmaf(rv.z, tv.z, hv.z);
        v.w = fmaf(rv.w, tv.w, hv.w);

        float* o = acc + (size_t)d * DIM + lane * 4;
        unsafeAtomicAdd(o + 0, v.x);
        unsafeAtomicAdd(o + 1, v.y);
        unsafeAtomicAdd(o + 2, v.z);
        unsafeAtomicAdd(o + 3, v.w);

        if (lane == 0) atomicAdd(indeg + d, 1);
    }
}

// ---------------------------------------------------------------------------
// Kernel 2: swizzle the three 128x128 fp32 weight matrices into the exact
// per-lane bf16 B-operand layout of v_wmma_f32_16x16x32_bf16 (ISA 7.12.2):
//   B element e of lane L (tile nt, kstep ks) = W[ks*32 + (L>>4)*8 + (e<8 ? e : e+8)][nt*16 + (L&15)]
// Flat index: (((mat*8 + nt)*4 + ks)*32 + lane)*16 + e
// ---------------------------------------------------------------------------
__global__ void weight_swizzle_kernel(const float* __restrict__ Wn,
                                      const float* __restrict__ Wl,
                                      const float* __restrict__ We,
                                      __bf16*      __restrict__ wsB)
{
    const int idx = blockIdx.x * blockDim.x + threadIdx.x;
    if (idx >= SWZ_ELEMS) return;
    const int e    =  idx        & 15;
    const int lane = (idx >> 4)  & 31;
    const int ks   = (idx >> 9)  & 3;
    const int nt   = (idx >> 11) & 7;
    const int mat  =  idx >> 14;

    const int n = nt * 16 + (lane & 15);
    const int k = ks * 32 + ((lane >> 4) * 8) + (e < 8 ? e : e + 8);
    const float* W = (mat == 0) ? Wn : ((mat == 1) ? Wl : We);
    wsB[idx] = (__bf16)W[k * DIM + n];
}

// ---------------------------------------------------------------------------
// A-operand pack: 16-bit A 16x32 layout (ISA 7.12.2): lane L (M = L&15) holds
// K = kb..kb+7 in elements 0..7 and K = kb+16..kb+23 in elements 8..15, where
// kb = ks*32 + (L>>4)*8.
// ---------------------------------------------------------------------------
__device__ __forceinline__ v16bf pack16(float4 a, float4 b, float4 c, float4 d)
{
    v16bf r;
    r[0]  = (__bf16)a.x;  r[1]  = (__bf16)a.y;  r[2]  = (__bf16)a.z;  r[3]  = (__bf16)a.w;
    r[4]  = (__bf16)b.x;  r[5]  = (__bf16)b.y;  r[6]  = (__bf16)b.z;  r[7]  = (__bf16)b.w;
    r[8]  = (__bf16)c.x;  r[9]  = (__bf16)c.y;  r[10] = (__bf16)c.z;  r[11] = (__bf16)c.w;
    r[12] = (__bf16)d.x;  r[13] = (__bf16)d.y;  r[14] = (__bf16)d.z;  r[15] = (__bf16)d.w;
    return r;
}

// ---------------------------------------------------------------------------
// Kernel 3: per-16-row-tile fused node GEMMs + epilogue.
//   out = relu( (acc @ Wn) * norm + h @ (indeg>0 ? Wl : We) )
// 256 threads = 8 waves per block. All 96 KB of swizzled bf16 weights are
// staged into LDS once per block (dynamic LDS; CDNA5 allows 320 KB/WGP), so
// the hot loop feeds WMMA from low-latency ds_load instead of VMEM.
// acc lives in d_out (in-place); A tiles are captured into registers BEFORE
// any stores, so the in-place update is safe. One wave per row tile.
// ---------------------------------------------------------------------------
__global__ void node_gemm_kernel(const float* __restrict__ h,
                                 const float* __restrict__ norm,
                                 const int*   __restrict__ indeg,
                                 const __bf16* __restrict__ wsB,
                                 float*       __restrict__ out,   // acc in / result out
                                 int N)
{
    extern __shared__ __bf16 ldsB[];              // SWZ_ELEMS bf16 = 96 KB

    // ---- cooperative stage: global (L2) -> LDS, 32B chunks, coalesced ----
    {
        const v16bf* g = (const v16bf*)wsB;
        v16bf*       l = (v16bf*)ldsB;
        for (int i = threadIdx.x; i < SWZ_VCHUNKS; i += blockDim.x)
            l[i] = g[i];
    }
    __syncthreads();                              // ALL waves reach this

    const int wave = blockIdx.x * (blockDim.x >> 5) + (threadIdx.x >> 5);
    const int lane = threadIdx.x & 31;
    const int ntiles = (N + 15) >> 4;
    if (wave >= ntiles) return;                   // wave-uniform, after barrier

    const int rowbase = wave * 16;
    const int m    = lane & 15;
    const int half = lane >> 4;                   // 0 or 1
    const int koff = half * 8;

    const int arow = min(rowbase + m, N - 1);
    const float* accRow = out + (size_t)arow * DIM;
    const float* hRow   = h   + (size_t)arow * DIM;

    // --- load & convert the two A tiles (acc rows, h rows) into registers ---
    v16bf aAcc[KSTEPS], aH[KSTEPS];
#pragma unroll
    for (int ks = 0; ks < KSTEPS; ++ks) {
        const int kb = ks * 32 + koff;
        float4 a0 = *(const float4*)(accRow + kb);
        float4 a1 = *(const float4*)(accRow + kb + 4);
        float4 a2 = *(const float4*)(accRow + kb + 16);
        float4 a3 = *(const float4*)(accRow + kb + 20);
        aAcc[ks]  = pack16(a0, a1, a2, a3);
        float4 b0 = *(const float4*)(hRow + kb);
        float4 b1 = *(const float4*)(hRow + kb + 4);
        float4 b2 = *(const float4*)(hRow + kb + 16);
        float4 b3 = *(const float4*)(hRow + kb + 20);
        aH[ks]    = pack16(b0, b1, b2, b3);
    }

    // per-lane epilogue scalars: rows rowbase + half*8 + {0..7}
    const int rb = min(rowbase + half * 8, max(N - 8, 0));
    float4 n0 = *(const float4*)(norm + rb);
    float4 n1 = *(const float4*)(norm + rb + 4);
    int4   g0 = *(const int4*)(indeg + rb);
    int4   g1 = *(const int4*)(indeg + rb + 4);
    const float nrm[8] = { n0.x, n0.y, n0.z, n0.w, n1.x, n1.y, n1.z, n1.w };
    const int   deg[8] = { g0.x, g0.y, g0.z, g0.w, g1.x, g1.y, g1.z, g1.w };

    const v16bf* B = (const v16bf*)ldsB;          // one v16bf == one lane's B slice

    if (rowbase + 16 <= N) {
        // ---------------- full-tile fast path (branch-free) ----------------
        for (int nt = 0; nt < NTILE_N; ++nt) {
            v8f cN = {}, cL = {}, cE = {};
#pragma unroll
            for (int ks = 0; ks < KSTEPS; ++ks) {
                v16bf bn = B[((0 * NTILE_N + nt) * KSTEPS + ks) * 32 + lane];
                v16bf bl = B[((1 * NTILE_N + nt) * KSTEPS + ks) * 32 + lane];
                v16bf be = B[((2 * NTILE_N + nt) * KSTEPS + ks) * 32 + lane];
                cN = __builtin_amdgcn_wmma_f32_16x16x32_bf16(false, aAcc[ks], false, bn,
                                                             (short)0, cN, false, false);
                cL = __builtin_amdgcn_wmma_f32_16x16x32_bf16(false, aH[ks],   false, bl,
                                                             (short)0, cL, false, false);
                cE = __builtin_amdgcn_wmma_f32_16x16x32_bf16(false, aH[ks],   false, be,
                                                             (short)0, cE, false, false);
            }
            // C/D layout: element r -> row = rowbase + half*8 + r, col = nt*16 + m
            const int col = nt * 16 + m;
#pragma unroll
            for (int r = 0; r < 8; ++r) {
                const int row = rowbase + half * 8 + r;
                float v = cN[r] * nrm[r] + (deg[r] > 0 ? cL[r] : cE[r]);
                out[(size_t)row * DIM + col] = fmaxf(v, 0.0f);
            }
        }
    } else {
        // ---------------- ragged last tile (guarded stores) ----------------
        for (int nt = 0; nt < NTILE_N; ++nt) {
            v8f cN = {}, cL = {}, cE = {};
#pragma unroll
            for (int ks = 0; ks < KSTEPS; ++ks) {
                v16bf bn = B[((0 * NTILE_N + nt) * KSTEPS + ks) * 32 + lane];
                v16bf bl = B[((1 * NTILE_N + nt) * KSTEPS + ks) * 32 + lane];
                v16bf be = B[((2 * NTILE_N + nt) * KSTEPS + ks) * 32 + lane];
                cN = __builtin_amdgcn_wmma_f32_16x16x32_bf16(false, aAcc[ks], false, bn,
                                                             (short)0, cN, false, false);
                cL = __builtin_amdgcn_wmma_f32_16x16x32_bf16(false, aH[ks],   false, bl,
                                                             (short)0, cL, false, false);
                cE = __builtin_amdgcn_wmma_f32_16x16x32_bf16(false, aH[ks],   false, be,
                                                             (short)0, cE, false, false);
            }
            const int col = nt * 16 + m;
#pragma unroll
            for (int r = 0; r < 8; ++r) {
                const int row = rowbase + half * 8 + r;
                if (row < N) {
                    float v = cN[r] * nrm[r] + (deg[r] > 0 ? cL[r] : cE[r]);
                    out[(size_t)row * DIM + col] = fmaxf(v, 0.0f);
                }
            }
        }
    }
}

// ---------------------------------------------------------------------------
extern "C" void kernel_launch(void* const* d_in, const int* in_sizes, int n_in,
                              void* d_out, int out_size, void* d_ws, size_t ws_size,
                              hipStream_t stream)
{
    const float* h    = (const float*)d_in[0];
    const float* rel  = (const float*)d_in[1];
    const float* tim  = (const float*)d_in[2];
    const float* Wn   = (const float*)d_in[3];
    const float* Wl   = (const float*)d_in[4];
    const float* We   = (const float*)d_in[5];
    const float* norm = (const float*)d_in[6];
    const int*   src  = (const int*)d_in[7];
    const int*   dst  = (const int*)d_in[8];
    const int*   etyp = (const int*)d_in[9];
    const int*   etim = (const int*)d_in[10];
    float* out = (float*)d_out;

    const int N = in_sizes[6];   // norm has N elements
    const int E = in_sizes[7];

    // workspace: [indeg: N ints][swizzled bf16 weights: 49152 * 2B]  (< 300 KB)
    int* indeg = (int*)d_ws;
    const size_t indegBytes = ((size_t)N * sizeof(int) + 255) & ~(size_t)255;
    __bf16* wsB = (__bf16*)((char*)d_ws + indegBytes);

    // d_out doubles as the fp32 edge accumulator -> must start at zero
    hipMemsetAsync(d_out, 0, (size_t)N * DIM * sizeof(float), stream);
    hipMemsetAsync(indeg, 0, (size_t)N * sizeof(int), stream);

    weight_swizzle_kernel<<<(SWZ_ELEMS + 255) / 256, 256, 0, stream>>>(Wn, Wl, We, wsB);

    edge_scatter_kernel<<<4096, 256, 0, stream>>>(h, rel, tim, src, dst, etyp, etim,
                                                  out, indeg, E);

    const int ntiles  = (N + 15) >> 4;           // 3125 row tiles
    const int gblocks = (ntiles + 7) / 8;        // 8 waves (256 threads) per block
    node_gemm_kernel<<<gblocks, 256, SWZ_ELEMS * sizeof(__bf16), stream>>>(
        h, norm, indeg, wsB, out, N);
}
